// POlogPolicyModel_49306224558584
// MI455X (gfx1250) — matrix operand order repaired and checked by reading, not compiled, not run on hardware
//
#include <hip/hip_runtime.h>

// ---------------------------------------------------------------------------
// POlogPolicyModel on MI455X (gfx1250): 32 paths / workgroup (two 16-row M
// tiles sharing each B fragment), dual-phase linear history buffer so layer-1
// A fragments load as ds_load_2addr_b32 with immediate offsets, fmaxf-based
// ReLU (single v_max_num_f32), 256-step scan, v_wmma_f32_16x16x32_bf16.
// ---------------------------------------------------------------------------

typedef __attribute__((ext_vector_type(16))) __bf16 v16bf;
typedef __attribute__((ext_vector_type(8)))  float  v8f;

#define NTC        256
#define DTC        0.04f
#define NHIST      64
#define HIDDEN     512
#define LAMBD      1.0f
#define MU1        0.08f
#define MU2        0.02f
#define RR         0.02f
#define SIGMA      0.2f
#define BETA       0.1f
#define ETA        0.5f
#define LOGEPS     1e-6f
#define UPEN       100.0f

#define MROWS      32       // paths per workgroup
#define H_STRIDE   520      // 512 + 8 bf16 pad (16B) -> banks spread for b128
#define HB_STRIDE  328      // history row stride (>= 64 + 256 + 1, padded)

__device__ __forceinline__ unsigned short f2bf16(float f) {
    unsigned int u = __float_as_uint(f);
    u += 0x7fffu + ((u >> 16) & 1u);          // round-to-nearest-even
    return (unsigned short)(u >> 16);
}

// B fragment: packed by pack_w into per-lane 32B chunks (WMMA B layout).
__device__ __forceinline__ v16bf load_bfrag(const unsigned int* wp, int block, int lane) {
    const uint4* p = (const uint4*)(wp + ((size_t)block * 32 + lane) * 8);
    v16bf b;
    *(uint4*)&b       = p[0];
    *((uint4*)&b + 1) = p[1];
    return b;
}

// A fragment from a row-major bf16 LDS activation tile (MROWS x 512).
// 16-bit A 16x32 layout: lane<16 holds k {0..7,16..23}, lane>=16 holds k {8..15,24..31}.
__device__ __forceinline__ v16bf load_afrag_h(const unsigned short* h, int kc, int lane, int rowoff) {
    int row   = (lane & 15) + rowoff;
    int khalf = (lane < 16) ? 0 : 8;
    const unsigned short* base = h + row * H_STRIDE + kc * 32 + khalf;
    v16bf a;
    *(uint4*)&a       = *(const uint4*)(base);
    *((uint4*)&a + 1) = *(const uint4*)(base + 16);
    return a;
}

// A fragment from the linear history buffer: window element k of step t lives
// at element offset tt + k (tt pre-adjusted for buffer phase, parity-matched)
// -> every VGPR pair (k,k+1) is one dword-aligned ds_load_b32 at an immediate
// offset from a single per-lane base address.
__device__ __forceinline__ v16bf load_afrag_hist(const unsigned short* hb, int tt, int kc,
                                                 int lane, int rowoff) {
    int row  = (lane & 15) + rowoff;
    int koff = (lane < 16) ? 0 : 8;
    const unsigned short* base = hb + row * HB_STRIDE + tt + kc * 32 + koff;
    union { unsigned int u[8]; v16bf v; } a;
#pragma unroll
    for (int j = 0; j < 8; ++j) {
        int kl = (j < 4) ? (2 * j) : (16 + 2 * (j - 4));
        a.u[j] = *(const unsigned int*)(base + kl);
    }
    return a.v;
}

__device__ __forceinline__ v8f wmma_bf16(v16bf a, v16bf b, v8f c) {
    return __builtin_amdgcn_wmma_f32_16x16x32_bf16(false, a, false, b, (short)0, c, false, false);
}

// ---------------------------------------------------------------------------
// Weight pre-pack: fp32 row-major (K x realN) -> bf16 WMMA B-fragment blocks.
// Block (nt, kc) = 32 lanes x 32 bytes; blockIdx.x = nt*num_kc + kc.
// ---------------------------------------------------------------------------
__global__ void pack_w(const float* __restrict__ W, unsigned int* __restrict__ dst,
                       int num_kc, int num_nt, int realN) {
    int lane = threadIdx.x;
    int blk  = blockIdx.x;
    int kc   = blk % num_kc;
    int nt   = blk / num_kc;
    int col  = nt * 16 + (lane & 15);
    int koff = kc * 32 + ((lane < 16) ? 0 : 8);
    unsigned int* d = dst + ((size_t)blk * 32 + lane) * 8;
#pragma unroll
    for (int j = 0; j < 8; ++j) {
        int   k0 = koff + ((j < 4) ? (2 * j) : (16 + 2 * (j - 4)));
        float lo = (col < realN) ? W[(size_t)k0 * realN + col]       : 0.f;
        float hi = (col < realN) ? W[(size_t)(k0 + 1) * realN + col] : 0.f;
        d[j] = ((unsigned)f2bf16(hi) << 16) | (unsigned)f2bf16(lo);
    }
}

// ---------------------------------------------------------------------------
// Main scan kernel: 32 paths per block, 128 threads = 4 waves.
// Wave w owns hidden columns [w*128, w*128+128) for BOTH 16-row M tiles.
// ---------------------------------------------------------------------------
__global__ __launch_bounds__(128) void polog_main(
    const float* __restrict__ dw, const float* __restrict__ x_init,
    const float* __restrict__ exp_array,
    const float* __restrict__ b1, const float* __restrict__ b2,
    const float* __restrict__ b3,
    const unsigned int* __restrict__ W1p, const unsigned int* __restrict__ W2p,
    const unsigned int* __restrict__ W3p, float* __restrict__ out) {

    __shared__ unsigned short s_h1[MROWS * H_STRIDE];
    __shared__ unsigned short s_h2[MROWS * H_STRIDE];
    __shared__ unsigned short s_hE[MROWS * HB_STRIDE];   // even-phase history
    __shared__ unsigned short s_hO[MROWS * HB_STRIDE];   // odd-phase (shifted +1)
    __shared__ float s_b1[HIDDEN];
    __shared__ float s_b2[HIDDEN];
    __shared__ float s_pi[MROWS * 2];

    const int tid  = threadIdx.x;
    const int lane = tid & 31;
    const int wave = tid >> 5;
    const int row0 = blockIdx.x * MROWS;

    for (int i = tid; i < HIDDEN; i += 128) { s_b1[i] = b1[i]; s_b2[i] = b2[i]; }
    for (int i = tid; i < MROWS * 64; i += 128) {
        int r = i >> 6, c = i & 63;
        unsigned short v = f2bf16(x_init[(size_t)(row0 + r) * 64 + c]);
        s_hE[r * HB_STRIDE + c]     = v;
        s_hO[r * HB_STRIDE + c + 1] = v;
    }

    // Per-path recurrent state lives in wave0 lanes 0..31 (one lane per path).
    float xs = 0.f, ys = 0.f, rw = 0.f;
    const float GEOM = (1.0f - __expf(-LAMBD * DTC * (float)(NHIST - 1))) / LAMBD;
    const float EDL  = __expf(-DTC * LAMBD);
    if (tid < MROWS) {
        const float* xi = x_init + (size_t)(row0 + tid) * 64;
        xs = xi[63];
        float acc = 0.f;
        for (int j = 1; j < 64; ++j) acc += exp_array[j - 1] * xi[j];
        ys = acc * DTC + GEOM * xi[0];
    }
    __syncthreads();

    for (int t = 0; t < NTC; ++t) {
        const unsigned short* hb = (t & 1) ? s_hO : s_hE;
        const int             tt = (t & 1) ? (t + 1) : t;

        // ---- Layer 1: (32x64) @ (64x512) ----
        v16bf a1[2][2];                     // [mh][kc]
        a1[0][0] = load_afrag_hist(hb, tt, 0, lane, 0);
        a1[0][1] = load_afrag_hist(hb, tt, 1, lane, 0);
        a1[1][0] = load_afrag_hist(hb, tt, 0, lane, 16);
        a1[1][1] = load_afrag_hist(hb, tt, 1, lane, 16);
#pragma unroll
        for (int nt = 0; nt < 8; ++nt) {
            int ntg = wave * 8 + nt;
            v8f accL = {0.f,0.f,0.f,0.f,0.f,0.f,0.f,0.f};
            v8f accH = {0.f,0.f,0.f,0.f,0.f,0.f,0.f,0.f};
#pragma unroll
            for (int kc = 0; kc < 2; ++kc) {
                v16bf bf = load_bfrag(W1p, ntg * 2 + kc, lane);
                accL = wmma_bf16(a1[0][kc], bf, accL);
                accH = wmma_bf16(a1[1][kc], bf, accH);
            }
            int   col  = ntg * 16 + (lane & 15);
            float bias = s_b1[col];
#pragma unroll
            for (int v = 0; v < 8; ++v) {
                int m = v + ((lane < 16) ? 0 : 8);
                s_h1[m * H_STRIDE + col]        = f2bf16(fmaxf(accL[v] + bias, 0.f));
                s_h1[(m + 16) * H_STRIDE + col] = f2bf16(fmaxf(accH[v] + bias, 0.f));
            }
        }
        __syncthreads();

        // ---- Layer 2: (32x512) @ (512x512) in two groups of 4 column tiles ----
#pragma unroll
        for (int g = 0; g < 2; ++g) {
            v8f accL[4], accH[4];
#pragma unroll
            for (int nt = 0; nt < 4; ++nt) {
                accL[nt] = (v8f){0.f,0.f,0.f,0.f,0.f,0.f,0.f,0.f};
                accH[nt] = (v8f){0.f,0.f,0.f,0.f,0.f,0.f,0.f,0.f};
            }
            for (int kc = 0; kc < 16; ++kc) {
                v16bf a2L = load_afrag_h(s_h1, kc, lane, 0);
                v16bf a2H = load_afrag_h(s_h1, kc, lane, 16);
#pragma unroll
                for (int nt = 0; nt < 4; ++nt) {
                    v16bf bf = load_bfrag(W2p, (wave * 8 + g * 4 + nt) * 16 + kc, lane);
                    accL[nt] = wmma_bf16(a2L, bf, accL[nt]);
                    accH[nt] = wmma_bf16(a2H, bf, accH[nt]);
                }
            }
#pragma unroll
            for (int nt = 0; nt < 4; ++nt) {
                int   col  = (wave * 8 + g * 4 + nt) * 16 + (lane & 15);
                float bias = s_b2[col];
#pragma unroll
                for (int v = 0; v < 8; ++v) {
                    int m = v + ((lane < 16) ? 0 : 8);
                    s_h2[m * H_STRIDE + col]        = f2bf16(fmaxf(accL[nt][v] + bias, 0.f));
                    s_h2[(m + 16) * H_STRIDE + col] = f2bf16(fmaxf(accH[nt][v] + bias, 0.f));
                }
            }
        }
        __syncthreads();

        // ---- Layer 3: (32x512) @ (512x16-padded); wave 0 -> rows 0..15, wave 1 -> 16..31 ----
        if (wave < 2) {
            int rowoff = wave * 16;
            v8f acc3 = {0.f,0.f,0.f,0.f,0.f,0.f,0.f,0.f};
            for (int kc = 0; kc < 16; ++kc)
                acc3 = wmma_bf16(load_afrag_h(s_h2, kc, lane, rowoff),
                                 load_bfrag(W3p, kc, lane), acc3);
            int n = lane & 15;
            if (n < 2) {
                float bias = b3[n];
#pragma unroll
                for (int v = 0; v < 8; ++v) {
                    int   m = v + ((lane < 16) ? 0 : 8) + rowoff;
                    float z = acc3[v] + bias;
                    s_pi[m * 2 + n] = 2.f / (1.f + __expf(-z));
                }
            }
        }
        __syncthreads();

        // ---- Scalar SDE update (lanes 0..31 of wave 0; one lane per path) ----
        if (tid < MROWS) {
            float pi0 = s_pi[tid * 2 + 0];
            float pi1 = s_pi[tid * 2 + 1];
            float disc  = __expf(-BETA * DTC * (float)t);
            float instr = __logf(fmaxf(pi0 * xs, 0.f) + LOGEPS) * disc
                        - fmaxf(-xs, 0.f) * UPEN;
            rw += instr * DTC;
            float dxv = ((MU1 - RR) * pi1 - pi0 + RR) * xs + MU2 * ys;
            float dwv = dw[(size_t)(row0 + tid) * NTC + t];
            float xn  = xs + dxv * DTC + SIGMA * xs * pi1 * dwv;
            ys = ys * EDL + xn * DTC;
            xs = xn;
            unsigned short xb = f2bf16(xn);          // history index 64+t
            s_hE[tid * HB_STRIDE + 64 + t] = xb;
            s_hO[tid * HB_STRIDE + 65 + t] = xb;
        }
        __syncthreads();
    }

    if (tid < MROWS) {
        const float FINAL_DISC = __expf(-BETA * (float)NTC * DTC);
        rw += __logf(fmaxf(xs + ETA * ys, 0.f) + LOGEPS) / BETA * FINAL_DISC;
        rw -= fmaxf(-xs, 0.f) * UPEN;
        out[row0 + tid] = -rw;
    }
}

// ---------------------------------------------------------------------------
extern "C" void kernel_launch(void* const* d_in, const int* in_sizes, int n_in,
                              void* d_out, int out_size, void* d_ws, size_t ws_size,
                              hipStream_t stream) {
    const float* dw      = (const float*)d_in[0];   // (16384, 256)
    const float* x_init  = (const float*)d_in[1];   // (16384, 64)
    const float* exp_arr = (const float*)d_in[2];   // (63,)
    const float* W1      = (const float*)d_in[3];   // (64, 512)
    const float* b1      = (const float*)d_in[4];   // (512,)
    const float* W2      = (const float*)d_in[5];   // (512, 512)
    const float* b2      = (const float*)d_in[6];   // (512,)
    const float* W3      = (const float*)d_in[7];   // (512, 2)
    const float* b3      = (const float*)d_in[8];   // (2,)
    float*       out     = (float*)d_out;

    // Workspace: packed bf16 weights in WMMA B-fragment blocks (1 KB / block).
    unsigned char* ws  = (unsigned char*)d_ws;
    unsigned int*  W1p = (unsigned int*)(ws);                          //  64 KB (64 blocks)
    unsigned int*  W2p = (unsigned int*)(ws + 64 * 1024);              // 512 KB (512 blocks)
    unsigned int*  W3p = (unsigned int*)(ws + 64 * 1024 + 512 * 1024); //  16 KB (16 blocks)

    pack_w<<<64,  32, 0, stream>>>(W1, W1p, /*num_kc=*/2,  /*num_nt=*/32, /*realN=*/512);
    pack_w<<<512, 32, 0, stream>>>(W2, W2p, /*num_kc=*/16, /*num_nt=*/32, /*realN=*/512);
    pack_w<<<16,  32, 0, stream>>>(W3, W3p, /*num_kc=*/16, /*num_nt=*/1,  /*realN=*/2);

    polog_main<<<16384 / MROWS, 128, 0, stream>>>(dw, x_init, exp_arr, b1, b2, b3,
                                                  W1p, W2p, W3p, out);
}
